// SpectralLoss_75436805587773
// MI455X (gfx1250) — matrix operand order repaired and checked by reading, not compiled, not run on hardware
//
#include <hip/hip_runtime.h>

#define BATCH 4096
#define FEAT  512
#define NDS   10000
#define GDIM  32      // 4096 / 128 block tile

typedef __attribute__((ext_vector_type(16))) __bf16 v16bf;
typedef __attribute__((ext_vector_type(8)))  __bf16 v8bf;
typedef __attribute__((ext_vector_type(4)))  __bf16 v4bf;
typedef __attribute__((ext_vector_type(8)))  float  v8f;

// Convert 8 + 8 consecutive f32 to a 16-element bf16 fragment (element order = load order).
__device__ __forceinline__ v16bf load_cvt_frag(const float* p0, const float* p1) {
  const float4 a0 = *(const float4*)(p0);
  const float4 a1 = *(const float4*)(p0 + 4);
  const float4 b0 = *(const float4*)(p1);
  const float4 b1 = *(const float4*)(p1 + 4);
  v16bf f;
  f[0]  = (__bf16)a0.x; f[1]  = (__bf16)a0.y; f[2]  = (__bf16)a0.z; f[3]  = (__bf16)a0.w;
  f[4]  = (__bf16)a1.x; f[5]  = (__bf16)a1.y; f[6]  = (__bf16)a1.z; f[7]  = (__bf16)a1.w;
  f[8]  = (__bf16)b0.x; f[9]  = (__bf16)b0.y; f[10] = (__bf16)b0.z; f[11] = (__bf16)b0.w;
  f[12] = (__bf16)b1.x; f[13] = (__bf16)b1.y; f[14] = (__bf16)b1.z; f[15] = (__bf16)b1.w;
  return f;
}

// 8 + 8 consecutive bf16 -> 16-element fragment (pure loads, no VALU).
__device__ __forceinline__ v16bf load_bf_frag(const __bf16* p0, const __bf16* p1) {
  const v8bf l0 = *(const v8bf*)(p0);
  const v8bf l1 = *(const v8bf*)(p1);
  return __builtin_shufflevector(l0, l1, 0, 1, 2, 3, 4, 5, 6, 7,
                                         8, 9, 10, 11, 12, 13, 14, 15);
}

// One wave per row: sq[row] = sum_k X[row,k]^2 (exact f32); optionally emit bf16 copy.
__global__ __launch_bounds__(256) void spectral_prep(const float* __restrict__ X,
                                                     float* __restrict__ sq,
                                                     __bf16* __restrict__ Xb,
                                                     int write_bf16) {
  const int wave = threadIdx.x >> 5;
  const int lane = threadIdx.x & 31;
  const int row  = blockIdx.x * 8 + wave;
  const float* rp = X + (size_t)row * FEAT;
  float s = 0.f;
#pragma unroll
  for (int t = 0; t < 4; ++t) {
    const int off = (lane << 2) + (t << 7);
    float4 v = *(const float4*)(rp + off);
    s += v.x * v.x + v.y * v.y + v.z * v.z + v.w * v.w;
    if (write_bf16) {
      v4bf b;
      b[0] = (__bf16)v.x; b[1] = (__bf16)v.y; b[2] = (__bf16)v.z; b[3] = (__bf16)v.w;
      *(v4bf*)(Xb + (size_t)row * FEAT + off) = b;
    }
  }
#pragma unroll
  for (int off = 16; off > 0; off >>= 1) s += __shfl_xor(s, off, 32);
  if (lane == 0) sq[row] = s;
}

// Shared epilogue: d2 = sq_i + sq_j - 2*gram; dist = d2>0 ? sqrt(d2) : 0;
// upper triangle only, weight (W[i,j] + W[j,i]); deterministic reduction to per-block partial.
__device__ __forceinline__ void epilogue(v8f acc[4][4], int gi0, int gj0,
                                         int laneLo, int laneHi, int lane, int wave,
                                         const float* __restrict__ adj,
                                         const int* __restrict__ bidx,
                                         const float* __restrict__ sq,
                                         float* __restrict__ part) {
  float local = 0.f;
#pragma unroll
  for (int ti = 0; ti < 4; ++ti) {
#pragma unroll
    for (int tj = 0; tj < 4; ++tj) {
      const int gj    = gj0 + tj * 16 + laneLo;
      const int bj    = bidx[gj];
      const float sqj = sq[gj];
      const int gib   = gi0 + ti * 16 + laneHi * 8;
      v8f c = acc[ti][tj];
#pragma unroll
      for (int r = 0; r < 8; ++r) {
        const int gi = gib + r;
        if (gj > gi) {
          float d2 = sq[gi] + sqj - 2.0f * c[r];
          if (d2 > 0.f) {
            const int bi_ = bidx[gi];
            const float w = adj[(size_t)bi_ * NDS + bj] + adj[(size_t)bj * NDS + bi_];
            local += w * __builtin_sqrtf(d2);
          }
        }
      }
    }
  }
#pragma unroll
  for (int off = 16; off > 0; off >>= 1) local += __shfl_xor(local, off, 32);

  __shared__ float wsum[4];
  if (lane == 0) wsum[wave] = local;
  __syncthreads();
  if (threadIdx.x == 0)
    part[blockIdx.y * GDIM + blockIdx.x] = wsum[0] + wsum[1] + wsum[2] + wsum[3];
}

// Main path: bf16-precomputed X -> inner loop is pure global_load_b128 + v_wmma.
__global__ __launch_bounds__(128) void spectral_gemm_pre(const __bf16* __restrict__ Xb,
                                                         const float* __restrict__ adj,
                                                         const int* __restrict__ bidx,
                                                         const float* __restrict__ sq,
                                                         float* __restrict__ part) {
  const int bx = blockIdx.x, by = blockIdx.y;
  if (bx < by) {
    if (threadIdx.x == 0) part[by * GDIM + bx] = 0.f;
    return;
  }
  const int wave   = threadIdx.x >> 5;
  const int lane   = threadIdx.x & 31;
  const int laneLo = lane & 15;
  const int laneHi = lane >> 4;
  const int gi0 = by * 128 + (wave >> 1) * 64;
  const int gj0 = bx * 128 + (wave & 1) * 64;

  v8f acc[4][4];
#pragma unroll
  for (int i = 0; i < 4; ++i)
#pragma unroll
    for (int j = 0; j < 4; ++j) acc[i][j] = (v8f){};

  for (int k0 = 0; k0 < FEAT; k0 += 32) {
    v16bf a[4], b[4];
#pragma unroll
    for (int t = 0; t < 4; ++t) {
      // A: lane = row m, K octets {0-7,16-23} (lanes 0-15) / {8-15,24-31} (lanes 16-31)
      const __bf16* ap = Xb + (size_t)(gi0 + t * 16 + laneLo) * FEAT + k0 + laneHi * 8;
      a[t] = load_bf_frag(ap, ap + 16);
      // B: lane = column n, K octets {0-7,16-23} (lanes 0-15) / {8-15,24-31} (lanes 16-31)
      const __bf16* bp = Xb + (size_t)(gj0 + t * 16 + laneLo) * FEAT + k0 + laneHi * 8;
      b[t] = load_bf_frag(bp, bp + 16);
    }
#pragma unroll
    for (int i = 0; i < 4; ++i)
#pragma unroll
      for (int j = 0; j < 4; ++j)
        acc[i][j] = __builtin_amdgcn_wmma_f32_16x16x32_bf16(
            false, a[i], false, b[j], (short)0, acc[i][j], false, false);
  }

  epilogue(acc, gi0, gj0, laneLo, laneHi, lane, wave, adj, bidx, sq, part);
}

// Fallback path (ws too small for bf16 copy): convert f32->bf16 on the fly.
__global__ __launch_bounds__(128) void spectral_gemm_fly(const float* __restrict__ X,
                                                         const float* __restrict__ adj,
                                                         const int* __restrict__ bidx,
                                                         const float* __restrict__ sq,
                                                         float* __restrict__ part) {
  const int bx = blockIdx.x, by = blockIdx.y;
  if (bx < by) {
    if (threadIdx.x == 0) part[by * GDIM + bx] = 0.f;
    return;
  }
  const int wave   = threadIdx.x >> 5;
  const int lane   = threadIdx.x & 31;
  const int laneLo = lane & 15;
  const int laneHi = lane >> 4;
  const int gi0 = by * 128 + (wave >> 1) * 64;
  const int gj0 = bx * 128 + (wave & 1) * 64;

  v8f acc[4][4];
#pragma unroll
  for (int i = 0; i < 4; ++i)
#pragma unroll
    for (int j = 0; j < 4; ++j) acc[i][j] = (v8f){};

  for (int k0 = 0; k0 < FEAT; k0 += 32) {
    v16bf a[4], b[4];
#pragma unroll
    for (int t = 0; t < 4; ++t) {
      const float* ap = X + (size_t)(gi0 + t * 16 + laneLo) * FEAT + k0 + laneHi * 8;
      a[t] = load_cvt_frag(ap, ap + 16);
      const float* bp = X + (size_t)(gj0 + t * 16 + laneLo) * FEAT + k0 + laneHi * 8;
      b[t] = load_cvt_frag(bp, bp + 16);
    }
#pragma unroll
    for (int i = 0; i < 4; ++i)
#pragma unroll
      for (int j = 0; j < 4; ++j)
        acc[i][j] = __builtin_amdgcn_wmma_f32_16x16x32_bf16(
            false, a[i], false, b[j], (short)0, acc[i][j], false, false);
  }

  epilogue(acc, gi0, gj0, laneLo, laneHi, lane, wave, adj, bidx, sq, part);
}

// Deterministic fixed-order final reduction of the 1024 block partials.
__global__ __launch_bounds__(256) void spectral_reduce(const float* __restrict__ part,
                                                       float* __restrict__ out) {
  __shared__ float s[256];
  float t = 0.f;
  for (int i = threadIdx.x; i < GDIM * GDIM; i += 256) t += part[i];
  s[threadIdx.x] = t;
  __syncthreads();
#pragma unroll
  for (int off = 128; off > 0; off >>= 1) {
    if (threadIdx.x < off) s[threadIdx.x] += s[threadIdx.x + off];
    __syncthreads();
  }
  if (threadIdx.x == 0)
    out[0] = s[0] * (1.0f / ((float)BATCH * (float)BATCH));
}

extern "C" void kernel_launch(void* const* d_in, const int* in_sizes, int n_in,
                              void* d_out, int out_size, void* d_ws, size_t ws_size,
                              hipStream_t stream) {
  const float* adj  = (const float*)d_in[0];   // [10000,10000] f32
  const float* X    = (const float*)d_in[1];   // [4096,512]    f32
  const int*   bidx = (const int*)d_in[2];     // [4096]        i32
  float* out = (float*)d_out;

  const size_t xb_bytes   = (size_t)BATCH * FEAT * sizeof(__bf16);   // 4 MB
  const size_t need_bytes = xb_bytes + (BATCH + GDIM * GDIM) * sizeof(float);
  const bool   use_pre    = (ws_size >= need_bytes);

  dim3 grid(GDIM, GDIM);
  if (use_pre) {
    __bf16* Xb  = (__bf16*)d_ws;
    float*  sq  = (float*)((char*)d_ws + xb_bytes);
    float*  part = sq + BATCH;
    spectral_prep<<<BATCH / 8, 256, 0, stream>>>(X, sq, Xb, 1);
    spectral_gemm_pre<<<grid, 128, 0, stream>>>(Xb, adj, bidx, sq, part);
    spectral_reduce<<<1, 256, 0, stream>>>(part, out);
  } else {
    float* sq   = (float*)d_ws;
    float* part = sq + BATCH;
    spectral_prep<<<BATCH / 8, 256, 0, stream>>>(X, sq, (__bf16*)nullptr, 0);
    spectral_gemm_fly<<<grid, 128, 0, stream>>>(X, adj, bidx, sq, part);
    spectral_reduce<<<1, 256, 0, stream>>>(part, out);
  }
}
